// Bidirectional_LSTM_Encoder_8349416423909
// MI455X (gfx1250) — matrix operand order repaired and checked by reading, not compiled, not run on hardware
//
#include <hip/hip_runtime.h>
#include <hip/hip_bf16.h>
#include <stdint.h>

// ---------------------------------------------------------------------------
// Bidirectional (feature-reversed) 2-layer LSTM for MI455X (gfx1250, wave32)
//   1) Input-side pre-gates via big parallel bf16 WMMA GEMMs (no recurrence).
//   2) Sequential scan does only the recurrent GEMM (K=512) fused with the
//      gate nonlinearities + c/h update (v_wmma_f32_16x16x32_bf16).
//   3) Weight tiles staged in LDS (ds_load) so each block reads W once.
// ---------------------------------------------------------------------------

typedef __attribute__((ext_vector_type(16))) __bf16 v16bf;
typedef __attribute__((ext_vector_type(8)))  __bf16 v8bf;
typedef __attribute__((ext_vector_type(8)))  float  v8f;

#define B_   128
#define T_   512
#define D_   512
#define H_   512
#define G4H_ 2048
#define MBIG ((size_t)B_ * T_)   // 65536
#define LDSROW (H_ + 16)         // 528 elems = 1056 B pitch (33*32B: aligned, bank-staggered)

// ---------------- WMMA helpers (layouts per CDNA5 ISA 7.12.2) ----------------

__device__ __forceinline__ v8f wmma_bf16(v16bf a, v16bf b, v8f c) {
  return __builtin_amdgcn_wmma_f32_16x16x32_bf16(
      /*neg_a=*/false, a, /*neg_b=*/false, b,
      /*c_mod=*/(short)0, c, /*reuse_a=*/false, /*reuse_b=*/false);
}

// A-matrix 16x32 bf16: lane (m = lane&15, hi = lane>>4) holds row m,
// elements 0..7  = K[8*hi .. +7], elements 8..15 = K[16+8*hi .. +7]
__device__ __forceinline__ v16bf load_a_tile(const __bf16* __restrict__ A, int lda,
                                             int m_lane, int hi, int k0) {
  const __bf16* pa = A + (size_t)m_lane * lda + k0 + 8 * hi;
  union { v16bf v; v8bf h[2]; } u;
  u.h[0] = *(const v8bf*)(pa);
  u.h[1] = *(const v8bf*)(pa + 16);
  return u.v;
}

// B-matrix 32x16 bf16 (B = W^T) from LDS-staged W rows: lane holds column
// n = lane&15 of B == ROW (g*16 + n) of the staged W,
// elements 0..7 = K[8*hi .. +7], elements 8..15 = K[16+8*hi .. +7].
__device__ __forceinline__ v16bf load_b_lds(const __bf16* lds, int ldsrow_idx,
                                            int hi, int k0) {
  const __bf16* pb = lds + (size_t)ldsrow_idx * LDSROW + k0 + 8 * hi;
  union { v16bf v; v8bf h[2]; } u;
  u.h[0] = *(const v8bf*)(pb);
  u.h[1] = *(const v8bf*)(pb + 16);
  return u.v;
}

__device__ __forceinline__ float fast_sigmoid(float x) {
  return 1.0f / (1.0f + __expf(-x));
}
__device__ __forceinline__ float fast_tanh(float x) {
  float ax = fabsf(x);
  float e  = __expf(2.0f * ax);            // inf for large ax -> t -> 1
  float t  = 1.0f - 2.0f / (e + 1.0f);
  return copysignf(t, x);
}

// ---------------- prep kernels ----------------

__global__ void f32_to_bf16_kernel(const float* __restrict__ src,
                                   __bf16* __restrict__ dst, size_t n) {
  size_t i = ((size_t)blockIdx.x * blockDim.x + threadIdx.x) * 4;
  if (i + 3 < n) {
    float4 v = *(const float4*)(src + i);
    dst[i + 0] = (__bf16)v.x; dst[i + 1] = (__bf16)v.y;
    dst[i + 2] = (__bf16)v.z; dst[i + 3] = (__bf16)v.w;
  } else {
    for (; i < n; ++i) dst[i] = (__bf16)src[i];
  }
}

// Convert weight (rows x cols) f32 -> bf16; optionally reverse columns
// (folds the reference's x[:, ::-1] into the 0b input weights).
__global__ void convert_weight_kernel(const float* __restrict__ src,
                                      __bf16* __restrict__ dst,
                                      int rows, int cols, int rev) {
  size_t i = (size_t)blockIdx.x * blockDim.x + threadIdx.x;
  size_t n = (size_t)rows * cols;
  if (i >= n) return;
  int r = (int)(i / cols), c = (int)(i % cols);
  int sc = rev ? (cols - 1 - c) : c;
  dst[(size_t)r * cols + c] = (__bf16)src[(size_t)r * cols + sc];
}

__global__ void combine_bias_kernel(const float* __restrict__ bih,
                                    const float* __restrict__ bhh,
                                    float* __restrict__ out, int n) {
  int i = blockIdx.x * blockDim.x + threadIdx.x;
  if (i < n) out[i] = bih[i] + bhh[i];
}

__global__ void zero_state_kernel(__bf16* __restrict__ h, float* __restrict__ c, int n) {
  int i = blockIdx.x * blockDim.x + threadIdx.x;
  if (i < n) { h[i] = (__bf16)0.0f; c[i] = 0.0f; }
}

// ---------------- big parallel pre-gate GEMM ----------------
// C[m][n] = sum_k A[m][k] * W[n][k] + bias[n], stored bf16.
// Block = 256 threads (8 waves). Block tile: 128 rows x 64 cols.
// The 64 needed W rows (n0 .. n0+63) are staged once into LDS.
__global__ __launch_bounds__(256)
void pregate_gemm_kernel(const __bf16* __restrict__ A,   // [M][K]
                         const __bf16* __restrict__ W,   // [N][K]
                         const float*  __restrict__ bias,// [N]
                         __bf16* __restrict__ C,         // [M][N]
                         int M, int N, int K) {
  __shared__ __bf16 ldsW[64 * LDSROW];                   // ~66 KB
  int wave   = threadIdx.x >> 5;
  int lane   = threadIdx.x & 31;
  int m_lane = lane & 15;
  int hi     = lane >> 4;
  int m0 = blockIdx.x * 128 + wave * 16;
  int n0 = blockIdx.y * 64;

  // cooperative stage: 64 rows x K bf16, 8-elem (16B) chunks
  for (int idx = threadIdx.x; idx < 64 * (H_ / 8); idx += 256) {
    int row = idx >> 6;                 // 0..63  (H_/8 == 64 chunks per row)
    int c8  = (idx & 63) << 3;
    *(v8bf*)(ldsW + (size_t)row * LDSROW + c8) =
        *(const v8bf*)(W + (size_t)(n0 + row) * K + c8);
  }
  __syncthreads();

  const __bf16* Arow = A + (size_t)m0 * K;
  v8f acc[4] = {};
  for (int k0 = 0; k0 < K; k0 += 32) {
    v16bf a = load_a_tile(Arow, K, m_lane, hi, k0);
#pragma unroll
    for (int g = 0; g < 4; ++g) {
      v16bf b = load_b_lds(ldsW, g * 16 + m_lane, hi, k0);
      acc[g] = wmma_bf16(a, b, acc[g]);
    }
  }
#pragma unroll
  for (int g = 0; g < 4; ++g) {
    int   n  = n0 + g * 16 + m_lane;
    float bs = bias[n];
#pragma unroll
    for (int r = 0; r < 8; ++r) {
      int m = m0 + 8 * hi + r;             // C layout: row = r + 8*hi
      C[(size_t)m * N + n] = (__bf16)(acc[g][r] + bs);
    }
  }
}

// ---------------- fused recurrent step ----------------
// grid = (32 hidden-unit tiles, 2 directions), block = 256 (8 waves).
// Wave w: batch rows b0 = 16*w. Each block owns 16 hidden units j for all 4
// gates (W rows g*H_ + j0 + n staged in LDS), so it fully updates c/h.
__global__ __launch_bounds__(256)
void lstm_step_kernel(const __bf16* __restrict__ h_prev,  // [2][B_][H_]
                      __bf16*       __restrict__ h_next,  // [2][B_][H_]
                      const float*  __restrict__ c_prev,  // [2][B_][H_]
                      float*        __restrict__ c_next,  // [2][B_][H_]
                      const __bf16* __restrict__ Whh,     // [2][G4H_][H_] (dir-major)
                      const __bf16* __restrict__ Gpre,    // + dir*g_dir + b*g_b + n
                      size_t g_dir, size_t g_b,
                      __bf16*       __restrict__ h_arch,  // layer0: + dir*a_dir + b*a_b + j
                      size_t a_dir, size_t a_b,
                      float*        __restrict__ out_h,   // layer1: + dir*o_dir + b*o_b + j
                      float*        __restrict__ out_c,
                      size_t o_dir, size_t o_b) {
  __shared__ __bf16 ldsW[64 * LDSROW];                    // ~66 KB
  int dir    = blockIdx.y;
  int j0     = blockIdx.x * 16;
  int wave   = threadIdx.x >> 5;
  int lane   = threadIdx.x & 31;
  int m_lane = lane & 15;
  int hi     = lane >> 4;
  int b0     = wave * 16;

  const __bf16* Wd = Whh + (size_t)dir * G4H_ * H_;
  // stage the 4 gates' 16 W rows each: ldsrow = g*16+n <- W row g*H_ + j0 + n
  for (int idx = threadIdx.x; idx < 64 * (H_ / 8); idx += 256) {
    int row = idx >> 6;
    int c8  = (idx & 63) << 3;
    int g   = row >> 4, n = row & 15;
    *(v8bf*)(ldsW + (size_t)row * LDSROW + c8) =
        *(const v8bf*)(Wd + (size_t)(g * H_ + j0 + n) * H_ + c8);
  }
  __syncthreads();

  const __bf16* Arow = h_prev + ((size_t)dir * B_ + b0) * H_;
  v8f acc[4] = {};
  for (int k0 = 0; k0 < H_; k0 += 32) {
    v16bf a = load_a_tile(Arow, H_, m_lane, hi, k0);
#pragma unroll
    for (int g = 0; g < 4; ++g) {
      v16bf b = load_b_lds(ldsW, g * 16 + m_lane, hi, k0);
      acc[g] = wmma_bf16(a, b, acc[g]);
    }
  }

  int j = j0 + m_lane;
#pragma unroll
  for (int r = 0; r < 8; ++r) {
    int    b     = b0 + 8 * hi + r;
    size_t gbase = (size_t)dir * g_dir + (size_t)b * g_b;
    float ig = acc[0][r] + (float)Gpre[gbase + 0 * H_ + j];
    float fg = acc[1][r] + (float)Gpre[gbase + 1 * H_ + j];
    float gg = acc[2][r] + (float)Gpre[gbase + 2 * H_ + j];
    float og = acc[3][r] + (float)Gpre[gbase + 3 * H_ + j];
    float iv = fast_sigmoid(ig);
    float fv = fast_sigmoid(fg);
    float gv = fast_tanh(gg);
    float ov = fast_sigmoid(og);
    size_t s = ((size_t)dir * B_ + b) * H_ + j;
    float cv = fv * c_prev[s] + iv * gv;
    float hv = ov * fast_tanh(cv);
    c_next[s] = cv;
    h_next[s] = (__bf16)hv;
    if (h_arch) h_arch[(size_t)dir * a_dir + (size_t)b * a_b + j] = (__bf16)hv;
    if (out_h) {
      size_t o = (size_t)dir * o_dir + (size_t)b * o_b + j;
      out_h[o] = hv;
      out_c[o] = cv;
    }
  }
}

// out[:, -1] and c_out[:, -1]
__global__ void final_copy_kernel(const float* __restrict__ outh,
                                  const float* __restrict__ outc,
                                  float* __restrict__ oh, float* __restrict__ oc) {
  int i = blockIdx.x * blockDim.x + threadIdx.x;
  if (i >= B_ * 2 * H_) return;
  int b = i >> 10, k = i & 1023;
  size_t src = ((size_t)b * T_ + (T_ - 1)) * (2 * H_) + k;
  oh[i] = outh[src];
  oc[i] = outc[src];
}

// ---------------------------------------------------------------------------

extern "C" void kernel_launch(void* const* d_in, const int* in_sizes, int n_in,
                              void* d_out, int out_size, void* d_ws, size_t ws_size,
                              hipStream_t stream) {
  (void)in_sizes; (void)n_in; (void)out_size; (void)ws_size;
  const float* in_x = (const float*)d_in[0];
  // cell order: 0 = 0f, 1 = 0b, 2 = 1f, 3 = 1b
  const float* w_ih[4] = {(const float*)d_in[1],  (const float*)d_in[5],
                          (const float*)d_in[9],  (const float*)d_in[13]};
  const float* w_hh[4] = {(const float*)d_in[2],  (const float*)d_in[6],
                          (const float*)d_in[10], (const float*)d_in[14]};
  const float* b_ih[4] = {(const float*)d_in[3],  (const float*)d_in[7],
                          (const float*)d_in[11], (const float*)d_in[15]};
  const float* b_hh[4] = {(const float*)d_in[4],  (const float*)d_in[8],
                          (const float*)d_in[12], (const float*)d_in[16]};

  // ---- carve workspace ----
  uint8_t* p = (uint8_t*)d_ws;
  auto carve = [&](size_t bytes) -> uint8_t* {
    uint8_t* r = p;
    p += (bytes + 255) & ~(size_t)255;
    return r;
  };
  const size_t WCNT = (size_t)G4H_ * H_;               // 2048*512 per matrix
  __bf16* Xbf   = (__bf16*)carve(MBIG * D_ * 2);       // 64 MB  input bf16
  __bf16* Wih   = (__bf16*)carve(4 * WCNT * 2);        // 8 MB
  __bf16* Whh   = (__bf16*)carve(4 * WCNT * 2);        // 8 MB  (dir-major per layer)
  float*  bias  = (float*) carve(4 * G4H_ * 4);        // 32 KB combined biases
  __bf16* Xg    = (__bf16*)carve(2 * MBIG * G4H_ * 2); // 512 MB pre-gates (reused)
  __bf16* h0a   = (__bf16*)carve(2 * MBIG * H_ * 2);   // 128 MB layer0 h history
  __bf16* hst0  = (__bf16*)carve(2 * B_ * H_ * 2);
  __bf16* hst1  = (__bf16*)carve(2 * B_ * H_ * 2);
  float*  cst0  = (float*) carve(2 * B_ * H_ * 4);
  float*  cst1  = (float*) carve(2 * B_ * H_ * 4);
  __bf16* hst[2] = {hst0, hst1};
  float*  cst[2] = {cst0, cst1};

  // ---- prep ----
  {
    size_t n = MBIG * D_;
    f32_to_bf16_kernel<<<(unsigned)((n / 4 + 255) / 256), 256, 0, stream>>>(in_x, Xbf, n);
  }
  for (int c = 0; c < 4; ++c) {
    unsigned blocks = (unsigned)((WCNT + 255) / 256);
    // cell 1 (= 0b) gets feature-reversed input -> reverse its W_ih columns
    convert_weight_kernel<<<blocks, 256, 0, stream>>>(w_ih[c], Wih + c * WCNT,
                                                      G4H_, H_, c == 1 ? 1 : 0);
    convert_weight_kernel<<<blocks, 256, 0, stream>>>(w_hh[c], Whh + c * WCNT,
                                                      G4H_, H_, 0);
    combine_bias_kernel<<<(G4H_ + 255) / 256, 256, 0, stream>>>(b_ih[c], b_hh[c],
                                                                bias + c * G4H_, G4H_);
  }

  const dim3 gemm_grid((unsigned)(MBIG / 128), G4H_ / 64);
  const size_t g_dir = MBIG * G4H_, g_b = (size_t)T_ * G4H_;
  const size_t a_dir = MBIG * H_,   a_b = (size_t)T_ * H_;
  const int    stN = 2 * B_ * H_;   // state element count

  // ================= layer 0 =================
  pregate_gemm_kernel<<<gemm_grid, 256, 0, stream>>>(Xbf, Wih + 0 * WCNT, bias + 0 * G4H_,
                                                     Xg, (int)MBIG, G4H_, D_);
  pregate_gemm_kernel<<<gemm_grid, 256, 0, stream>>>(Xbf, Wih + 1 * WCNT, bias + 1 * G4H_,
                                                     Xg + MBIG * G4H_, (int)MBIG, G4H_, D_);
  zero_state_kernel<<<(stN + 255) / 256, 256, 0, stream>>>(hst[0], cst[0], stN);

  for (int t = 0; t < T_; ++t) {
    int pp = t & 1;
    lstm_step_kernel<<<dim3(H_ / 16, 2), 256, 0, stream>>>(
        hst[pp], hst[1 - pp], cst[pp], cst[1 - pp],
        Whh + 0 * WCNT,                       // cells 0f,0b are dir-major contiguous
        Xg + (size_t)t * G4H_, g_dir, g_b,
        h0a + (size_t)t * H_, a_dir, a_b,     // archive h for layer-1 input GEMM
        nullptr, nullptr, 0, 0);
  }

  // ================= layer 1 =================
  pregate_gemm_kernel<<<gemm_grid, 256, 0, stream>>>(h0a, Wih + 2 * WCNT, bias + 2 * G4H_,
                                                     Xg, (int)MBIG, G4H_, H_);
  pregate_gemm_kernel<<<gemm_grid, 256, 0, stream>>>(h0a + MBIG * H_, Wih + 3 * WCNT,
                                                     bias + 3 * G4H_,
                                                     Xg + MBIG * G4H_, (int)MBIG, G4H_, H_);
  zero_state_kernel<<<(stN + 255) / 256, 256, 0, stream>>>(hst[0], cst[0], stN);

  float* out_h = (float*)d_out;                       // (B,T,2H)
  float* out_c = out_h + MBIG * (2 * H_);             // (B,T,2H)
  for (int t = 0; t < T_; ++t) {
    int pp = t & 1;
    lstm_step_kernel<<<dim3(H_ / 16, 2), 256, 0, stream>>>(
        hst[pp], hst[1 - pp], cst[pp], cst[1 - pp],
        Whh + 2 * WCNT,
        Xg + (size_t)t * G4H_, g_dir, g_b,
        nullptr, 0, 0,
        out_h + (size_t)t * (2 * H_), out_c + (size_t)t * (2 * H_),
        /*o_dir=*/(size_t)H_, /*o_b=*/(size_t)T_ * (2 * H_));
  }

  // final states out[:, -1], c_out[:, -1]
  float* oh = out_c + MBIG * (2 * H_);
  float* oc = oh + (size_t)B_ * (2 * H_);
  final_copy_kernel<<<(B_ * 2 * H_ + 255) / 256, 256, 0, stream>>>(out_h, out_c, oh, oc);
}
